// NequIPMessagePassingLayer_27573690040904
// MI455X (gfx1250) — matrix operand order repaired and checked by reading, not compiled, not run on hardware
//
#include <hip/hip_runtime.h>
#include <hip/hip_bf16.h>
#include <math.h>

typedef __attribute__((ext_vector_type(16))) _Float16 v16h;
typedef __attribute__((ext_vector_type(8)))  float    v8f;

#define CUTOFF_F 5.0f

// ---------------------------------------------------------------------------
// WMMA fragment helpers for V_WMMA_F32_16X16X32_F16 (gfx1250, wave32).
// A (16x32 f16): lane<16 -> M=lane, K = {0..7, 16..23}; lane>=16 -> K = {8..15, 24..31}
// B (32x16 f16): lane<16 -> N=lane, K = {0..7, 16..23}; lane>=16 -> K = {8..15, 24..31}
// C/D (16x16 f32): VGPR i -> M = i + 8*(lane>=16), N = lane&15
// ---------------------------------------------------------------------------

// A-fragment, contiguous K (k_stride==1): 4x b128 loads + 8 packed cvt.
__device__ __forceinline__ v16h frag_a_c(const float* __restrict__ src,
                                         int row_stride, int lane) {
  const int m    = lane & 15;
  const int koff = (lane >> 4) << 3;     // 0 or 8
  const float* p = src + (size_t)m * row_stride + koff;
  const float4 q0 = *(const float4*)(p + 0);
  const float4 q1 = *(const float4*)(p + 4);
  const float4 q2 = *(const float4*)(p + 16);
  const float4 q3 = *(const float4*)(p + 20);
  v16h f;
  f[0]  = (_Float16)q0.x; f[1]  = (_Float16)q0.y; f[2]  = (_Float16)q0.z; f[3]  = (_Float16)q0.w;
  f[4]  = (_Float16)q1.x; f[5]  = (_Float16)q1.y; f[6]  = (_Float16)q1.z; f[7]  = (_Float16)q1.w;
  f[8]  = (_Float16)q2.x; f[9]  = (_Float16)q2.y; f[10] = (_Float16)q2.z; f[11] = (_Float16)q2.w;
  f[12] = (_Float16)q3.x; f[13] = (_Float16)q3.y; f[14] = (_Float16)q3.z; f[15] = (_Float16)q3.w;
  return f;
}

// A-fragment, strided K (vector-channel streams, k_stride==3).
__device__ __forceinline__ v16h frag_a_s(const float* __restrict__ src,
                                         int row_stride, int k_stride, int lane) {
  const int m    = lane & 15;
  const int koff = (lane >> 4) << 3;
  const float* p = src + (size_t)m * row_stride;
  v16h f;
#pragma unroll
  for (int j = 0; j < 8; ++j) f[j]     = (_Float16)p[(size_t)(koff + j) * k_stride];
#pragma unroll
  for (int j = 0; j < 8; ++j) f[8 + j] = (_Float16)p[(size_t)(16 + koff + j) * k_stride];
  return f;
}

// B-fragment from the prepacked (f16, pre-scaled, register-ordered) buffer:
// frag fidx, lane -> 16 contiguous halves (32 B) => two global_load_b128.
__device__ __forceinline__ v16h frag_bp(const _Float16* __restrict__ base,
                                        int fidx, int lane) {
  return *(const v16h*)(base + ((size_t)fidx * 32 + lane) * 16);
}

__device__ __forceinline__ v8f wmma_f16(v16h a, v16h b, v8f c) {
  return __builtin_amdgcn_wmma_f32_16x16x32_f16(false, a, false, b,
                                                (short)0, c, false, false);
}

__device__ __forceinline__ float silu(float v) { return v / (1.0f + expf(-v)); }

// ---------------------------------------------------------------------------
// CSR builder: per-node edge counts -> exclusive scan -> bucket fill.
// Turns the segment_sum scatter (90M f32 atomics) into a per-node gather.
// ---------------------------------------------------------------------------
__global__ void __launch_bounds__(256)
count_kernel(const int* __restrict__ eidx, int* __restrict__ deg, int E) {
  const int e = blockIdx.x * 256 + threadIdx.x;
  if (e < E) atomicAdd(&deg[eidx[e]], 1);
}

__global__ void __launch_bounds__(256)
scan_kernel(const int* __restrict__ deg, int* __restrict__ offs, int N) {
  __shared__ int part[256];
  const int tid   = threadIdx.x;
  const int chunk = (N + 255) / 256;
  const int lo    = tid * chunk;
  const int hi    = min(N, lo + chunk);
  int sum = 0;
  for (int i = lo; i < hi; ++i) sum += deg[i];
  part[tid] = sum;
  __syncthreads();
#pragma unroll
  for (int off = 1; off < 256; off <<= 1) {   // Hillis-Steele inclusive scan
    int v = (tid >= off) ? part[tid - off] : 0;
    __syncthreads();
    if (tid >= off) part[tid] += v;
    __syncthreads();
  }
  int run = (tid == 0) ? 0 : part[tid - 1];
  for (int i = lo; i < hi; ++i) { offs[i] = run; run += deg[i]; }
  if (tid == 255) offs[N] = part[255];
}

__global__ void __launch_bounds__(256)
fill_kernel(const int* __restrict__ eidx, const int* __restrict__ offs,
            int* __restrict__ cur, int* __restrict__ elist, int E) {
  const int e = blockIdx.x * 256 + threadIdx.x;
  if (e < E) {
    const int c = eidx[e];
    const int p = atomicAdd(&cur[c], 1);
    elist[offs[c] + p] = e;
  }
}

// ---------------------------------------------------------------------------
// Kernel 0: repack a row-major f32 weight matrix (K x ncols) into B-fragment
// order, f16, with the GEMM scale factor folded in.  One wave per fragment.
// ---------------------------------------------------------------------------
__global__ void __launch_bounds__(128)
repack_b_kernel(const float* __restrict__ W, _Float16* __restrict__ dst,
                int ld, int ntiles, int kchunks, float scale) {
  const int lane = threadIdx.x & 31;
  const int f = blockIdx.x * 4 + (threadIdx.x >> 5);
  if (f >= ntiles * kchunks) return;
  const int nt = f / kchunks;
  const int kc = f % kchunks;
  const int n  = lane & 15;
  const int k0 = kc * 32 + ((lane >> 4) << 3);
  _Float16* o = dst + ((size_t)f * 32 + lane) * 16;
#pragma unroll
  for (int j = 0; j < 16; ++j)
    o[j] = (_Float16)(W[(size_t)(k0 + (j & 7) + ((j >> 3) << 4)) * ld + nt * 16 + n] * scale);
}

// ---------------------------------------------------------------------------
// Kernel 1: node pre-mix.  h0 = x0 @ W_pre0 * inv_c ; h1[:,:,x] = x1[:,:,x] @ W_pre1 * inv_c
// One wave per 16-node tile.  h1 stored as (N, 64, 3).
// ---------------------------------------------------------------------------
__global__ void __launch_bounds__(128)
pre_kernel(const float* __restrict__ node, const _Float16* __restrict__ Wp0f,
           const _Float16* __restrict__ Wp1f, float* __restrict__ h0,
           float* __restrict__ h1, int N) {
  const int lane = threadIdx.x & 31;
  const int tile = blockIdx.x * 4 + (threadIdx.x >> 5);
  if (tile * 16 >= N) return;               // wave-uniform: EXEC stays all-ones
  const int r0 = tile * 16;
  const int mh = (lane >> 4) * 8, n = lane & 15;

  { // scalar stream: x0 @ W_pre0
    v16h a0 = frag_a_c(node + (size_t)r0 * 256 + 0,  256, lane);
    v16h a1 = frag_a_c(node + (size_t)r0 * 256 + 32, 256, lane);
#pragma unroll
    for (int nt = 0; nt < 4; ++nt) {
      v8f acc = {};
      acc = wmma_f16(a0, frag_bp(Wp0f, nt * 2 + 0, lane), acc);
      acc = wmma_f16(a1, frag_bp(Wp0f, nt * 2 + 1, lane), acc);
      float* dst = h0 + (size_t)r0 * 64 + nt * 16 + n;
#pragma unroll
      for (int i = 0; i < 8; ++i) dst[(size_t)(mh + i) * 64] = acc[i];
    }
  }
#pragma unroll
  for (int x = 0; x < 3; ++x) { // vector streams: x1[:,:,x] @ W_pre1
    v16h a0 = frag_a_s(node + (size_t)r0 * 256 + 64 + x,          256, 3, lane);
    v16h a1 = frag_a_s(node + (size_t)r0 * 256 + 64 + x + 3 * 32, 256, 3, lane);
#pragma unroll
    for (int nt = 0; nt < 4; ++nt) {
      v8f acc = {};
      acc = wmma_f16(a0, frag_bp(Wp1f, nt * 2 + 0, lane), acc);
      acc = wmma_f16(a1, frag_bp(Wp1f, nt * 2 + 1, lane), acc);
      float* dst = h1 + (size_t)r0 * 192 + (nt * 16 + n) * 3 + x;
#pragma unroll
      for (int i = 0; i < 8; ++i) dst[(size_t)(mh + i) * 192] = acc[i];
    }
  }
}

// ---------------------------------------------------------------------------
// Kernel 2: per-node gather.  One wave per destination node walks its CSR edge
// list, recomputes each edge's radial MLP once (register-resident), and
// accumulates all 704 message slots in registers.  Zero float atomics.
// Each lane owns channels {lane, lane+32}.
// s0: (N,128) [mA|mD], s1o: (N,128,3) [mB|mC], s1e: (N,64,3) [mE]
// ---------------------------------------------------------------------------
__global__ void __launch_bounds__(256)
node_gather_kernel(const float* __restrict__ dist, const float* __restrict__ eemb,
                   const int* __restrict__ eidx, const float* __restrict__ freqs,
                   const float* __restrict__ W1, const float* __restrict__ b1,
                   const float* __restrict__ W2, const float* __restrict__ b2,
                   const float* __restrict__ W3, const float* __restrict__ b3,
                   const float* __restrict__ h0, const float* __restrict__ h1,
                   const int* __restrict__ offs, const int* __restrict__ elist,
                   float* __restrict__ s0, float* __restrict__ s1o,
                   float* __restrict__ s1e, int N, int E) {
  const int lane = threadIdx.x & 31;
  const int node = blockIdx.x * 8 + (threadIdx.x >> 5);
  if (node >= N) return;                     // wave-uniform
  const int beg = offs[node];
  const int end = offs[node + 1];

  float a0[2] = {0.f, 0.f}, aD[2] = {0.f, 0.f};
  float aB[2][3] = {}, aC[2][3] = {}, aE[2][3] = {};
  const float inv_s3 = 0.5773502691896258f;  // 1/sqrt(3)
  const float inv_s2 = 0.7071067811865476f;  // 1/sqrt(2)

  for (int idx = beg; idx < end; ++idx) {
    const int e   = elist[idx];
    const int nbr = eidx[E + e];
    // Hide the gather latency behind the radial MLP (global_prefetch_b8).
    __builtin_prefetch(h0 + (size_t)nbr * 64 + lane, 0, 3);
    __builtin_prefetch(h1 + (size_t)nbr * 192 + lane * 3, 0, 3);
    __builtin_prefetch(h1 + (size_t)nbr * 192 + 96 + lane * 3, 0, 3);

    const float r = dist[e];
    const float d = r / CUTOFF_F;
    const float d2 = d * d, d6 = d2 * d2 * d2, d7 = d6 * d, d8 = d7 * d;
    const float env = (d < 1.0f) ? (1.0f - 28.0f * d6 + 48.0f * d7 - 21.0f * d8) : 0.0f;
    const float pref = sqrtf(2.0f / CUTOFF_F) / r * env;

    float feat[8];
#pragma unroll
    for (int i = 0; i < 8; ++i) feat[i] = pref * sinf(freqs[i] * r / CUTOFF_F);

    float hA[10];
#pragma unroll
    for (int j = 0; j < 10; ++j) {
      float v = b1[j];
#pragma unroll
      for (int i = 0; i < 8; ++i) v += feat[i] * W1[i * 10 + j];
      hA[j] = silu(v);
    }
    float hB[10];
#pragma unroll
    for (int j = 0; j < 10; ++j) {
      float v = b2[j];
#pragma unroll
      for (int k = 0; k < 10; ++k) v += hA[k] * W2[k * 10 + j];
      hB[j] = silu(v);
    }

    const float e0  = eemb[(size_t)e * 4 + 0];
    const float e1x = eemb[(size_t)e * 4 + 1];
    const float e1y = eemb[(size_t)e * 4 + 2];
    const float e1z = eemb[(size_t)e * 4 + 3];

#pragma unroll
    for (int cc = 0; cc < 2; ++cc) {
      const int c = lane + cc * 32;
      float w[5];
#pragma unroll
      for (int g = 0; g < 5; ++g) {
        float v = b3[g * 64 + c];
#pragma unroll
        for (int j = 0; j < 10; ++j) v += hB[j] * W3[j * 320 + g * 64 + c];
        w[g] = v;
      }
      const float g0  = h0[(size_t)nbr * 64 + c];
      const float g1x = h1[(size_t)nbr * 192 + c * 3 + 0];
      const float g1y = h1[(size_t)nbr * 192 + c * 3 + 1];
      const float g1z = h1[(size_t)nbr * 192 + c * 3 + 2];

      a0[cc] += w[0] * g0 * e0;                                            // mA
      aD[cc] += w[3] * (g1x * e1x + g1y * e1y + g1z * e1z) * inv_s3;       // mD
      const float wbg = w[1] * g0;                                         // mB
      aB[cc][0] += wbg * e1x;
      aB[cc][1] += wbg * e1y;
      aB[cc][2] += wbg * e1z;
      const float wce = w[2] * e0;                                         // mC
      aC[cc][0] += wce * g1x;
      aC[cc][1] += wce * g1y;
      aC[cc][2] += wce * g1z;
      const float we = w[4] * inv_s2;                                      // mE
      aE[cc][0] += we * (g1y * e1z - g1z * e1y);
      aE[cc][1] += we * (g1z * e1x - g1x * e1z);
      aE[cc][2] += we * (g1x * e1y - g1y * e1x);
    }
  }

#pragma unroll
  for (int cc = 0; cc < 2; ++cc) {
    const int c = lane + cc * 32;
    s0[(size_t)node * 128 + c]      = a0[cc];
    s0[(size_t)node * 128 + 64 + c] = aD[cc];
#pragma unroll
    for (int x = 0; x < 3; ++x) {
      s1o[(size_t)node * 384 + c * 3 + x]        = aB[cc][x];
      s1o[(size_t)node * 384 + (64 + c) * 3 + x] = aC[cc][x];
      s1e[(size_t)node * 192 + c * 3 + x]        = aE[cc][x];
    }
  }
}

// ---------------------------------------------------------------------------
// Kernel 3: node post-mix + SiLU gating.  One wave per 16-node tile.
// Gates are kept in D-layout registers: y1o/y1e output tiles share the exact
// lane<->(M,N) mapping with gate tiles, so gating is a register multiply.
// ---------------------------------------------------------------------------
__global__ void __launch_bounds__(128)
post_kernel(const float* __restrict__ node, const float* __restrict__ s0,
            const float* __restrict__ s1o, const float* __restrict__ s1e,
            const _Float16* __restrict__ Wpost0f, const _Float16* __restrict__ Wpost1of,
            const _Float16* __restrict__ Wpost1ef, const _Float16* __restrict__ Wsi0f,
            const _Float16* __restrict__ Wsi1of, float* __restrict__ out, int N) {
  const int lane = threadIdx.x & 31;
  const int tile = blockIdx.x * 4 + (threadIdx.x >> 5);
  if (tile * 16 >= N) return;               // wave-uniform
  const int r0 = tile * 16;
  const int mh = (lane >> 4) * 8, n = lane & 15;

  // Phase 1: y0 = s0 @ W_post0 * inv_2c + x0 @ W_si0 * inv_c  (16 x 192)
  v16h as0[4], ax0[2];
#pragma unroll
  for (int k = 0; k < 4; ++k) as0[k] = frag_a_c(s0 + (size_t)r0 * 128 + k * 32, 128, lane);
#pragma unroll
  for (int k = 0; k < 2; ++k) ax0[k] = frag_a_c(node + (size_t)r0 * 256 + k * 32, 256, lane);

  v8f gacc[8];                               // silu(y0[:,64:192]) in D-layout
#pragma unroll
  for (int nt = 0; nt < 12; ++nt) {
    v8f acc = {};
#pragma unroll
    for (int k = 0; k < 4; ++k)
      acc = wmma_f16(as0[k], frag_bp(Wpost0f, nt * 4 + k, lane), acc);
#pragma unroll
    for (int k = 0; k < 2; ++k)
      acc = wmma_f16(ax0[k], frag_bp(Wsi0f, nt * 2 + k, lane), acc);
#pragma unroll
    for (int i = 0; i < 8; ++i) acc[i] = silu(acc[i]);
    if (nt < 4) {                            // scal -> out[:, 0:64]
      float* dst = out + (size_t)r0 * 448 + nt * 16 + n;
#pragma unroll
      for (int i = 0; i < 8; ++i) dst[(size_t)(mh + i) * 448] = acc[i];
    } else {
      gacc[nt - 4] = acc;                    // gates
    }
  }

  // Phase 2/3: y1o (gated by gacc[0..3]) and y1e (gated by gacc[4..7])
#pragma unroll
  for (int x = 0; x < 3; ++x) {
    v16h a1[4], axv[2], ae[2];
#pragma unroll
    for (int k = 0; k < 4; ++k)
      a1[k] = frag_a_s(s1o + (size_t)r0 * 384 + x + 3 * (k * 32), 384, 3, lane);
#pragma unroll
    for (int k = 0; k < 2; ++k)
      axv[k] = frag_a_s(node + (size_t)r0 * 256 + 64 + x + 3 * (k * 32), 256, 3, lane);
#pragma unroll
    for (int k = 0; k < 2; ++k)
      ae[k] = frag_a_s(s1e + (size_t)r0 * 192 + x + 3 * (k * 32), 192, 3, lane);

#pragma unroll
    for (int nt = 0; nt < 4; ++nt) {         // y1o -> out[:, 64 + c*3 + x]
      v8f acc = {};
#pragma unroll
      for (int k = 0; k < 4; ++k)
        acc = wmma_f16(a1[k], frag_bp(Wpost1of, nt * 4 + k, lane), acc);
#pragma unroll
      for (int k = 0; k < 2; ++k)
        acc = wmma_f16(axv[k], frag_bp(Wsi1of, nt * 2 + k, lane), acc);
      float* dst = out + (size_t)r0 * 448 + 64 + (nt * 16 + n) * 3 + x;
#pragma unroll
      for (int i = 0; i < 8; ++i) dst[(size_t)(mh + i) * 448] = acc[i] * gacc[nt][i];
    }
#pragma unroll
    for (int nt = 0; nt < 4; ++nt) {         // y1e -> out[:, 256 + c*3 + x]
      v8f acc = {};
#pragma unroll
      for (int k = 0; k < 2; ++k)
        acc = wmma_f16(ae[k], frag_bp(Wpost1ef, nt * 2 + k, lane), acc);
      float* dst = out + (size_t)r0 * 448 + 256 + (nt * 16 + n) * 3 + x;
#pragma unroll
      for (int i = 0; i < 8; ++i) dst[(size_t)(mh + i) * 448] = acc[i] * gacc[4 + nt][i];
    }
  }
}

// ---------------------------------------------------------------------------
extern "C" void kernel_launch(void* const* d_in, const int* in_sizes, int n_in,
                              void* d_out, int out_size, void* d_ws, size_t ws_size,
                              hipStream_t stream) {
  const float* node    = (const float*)d_in[0];
  // d_in[1] = Z_embeddings: unused by the reference
  const float* dist    = (const float*)d_in[2];
  const float* eemb    = (const float*)d_in[3];
  const int*   eidx    = (const int*)  d_in[4];
  const float* Wp0     = (const float*)d_in[5];
  const float* Wp1     = (const float*)d_in[6];
  const float* freqs   = (const float*)d_in[7];
  const float* W1      = (const float*)d_in[8];
  const float* b1      = (const float*)d_in[9];
  const float* W2      = (const float*)d_in[10];
  const float* b2      = (const float*)d_in[11];
  const float* W3      = (const float*)d_in[12];
  const float* b3      = (const float*)d_in[13];
  const float* Wpost0  = (const float*)d_in[14];
  const float* Wpost1o = (const float*)d_in[15];
  const float* Wpost1e = (const float*)d_in[16];
  const float* Wsi0    = (const float*)d_in[17];
  const float* Wsi1o   = (const float*)d_in[18];

  const int N = in_sizes[0] / 256;   // 8000 (multiple of 16)
  const int E = in_sizes[2];         // 128000

  const float inv_c  = 0.125f;                // 1/sqrt(64)
  const float inv_2c = 0.08838834764831845f;  // 1/sqrt(128)

  float* ws  = (float*)d_ws;         // layout: h0 | h1 | s0 | s1o | s1e | wfrags | ints
  float* h0  = ws;
  float* h1  = h0  + (size_t)N * 64;
  float* s0  = h1  + (size_t)N * 192;
  float* s1o = s0  + (size_t)N * 128;
  float* s1e = s1o + (size_t)N * 384;
  _Float16* wf = (_Float16*)(s1e + (size_t)N * 192);  // 32B-aligned region
  _Float16* wp0f     = wf;                  //  8 frags
  _Float16* wp1f     = wp0f     + 8  * 512; //  8 frags
  _Float16* wpost0f  = wp1f     + 8  * 512; // 48 frags
  _Float16* wsi0f    = wpost0f  + 48 * 512; // 24 frags
  _Float16* wpost1of = wsi0f    + 24 * 512; // 16 frags
  _Float16* wsi1of   = wpost1of + 16 * 512; //  8 frags
  _Float16* wpost1ef = wsi1of   + 8  * 512; //  8 frags
  int* ideg  = (int*)(wpost1ef + 8 * 512);  // N counts
  int* icur  = ideg + N;                    // N cursors
  int* ioffs = icur + N;                    // N+1 offsets
  int* ielist = ioffs + (N + 1);            // E edge ids

  // Zero the CSR counters (graph-capture-safe stream op); deg|cur adjacent.
  (void)hipMemsetAsync(ideg, 0, (size_t)2 * N * sizeof(int), stream);

  // Build CSR adjacency of edges grouped by destination node (ctr).
  count_kernel<<<(E + 255) / 256, 256, 0, stream>>>(eidx, ideg, E);
  scan_kernel<<<1, 256, 0, stream>>>(ideg, ioffs, N);
  fill_kernel<<<(E + 255) / 256, 256, 0, stream>>>(eidx, ioffs, icur, ielist, E);

  // One-shot weight repack into f16 B-fragment order (scales folded in).
  repack_b_kernel<<<2, 128, 0, stream>>>(Wp0,     wp0f,     64,  4,  2, inv_c);
  repack_b_kernel<<<2, 128, 0, stream>>>(Wp1,     wp1f,     64,  4,  2, inv_c);
  repack_b_kernel<<<12, 128, 0, stream>>>(Wpost0, wpost0f,  192, 12, 4, inv_2c);
  repack_b_kernel<<<6, 128, 0, stream>>>(Wsi0,    wsi0f,    192, 12, 2, inv_c);
  repack_b_kernel<<<4, 128, 0, stream>>>(Wpost1o, wpost1of, 64,  4,  4, inv_2c);
  repack_b_kernel<<<2, 128, 0, stream>>>(Wsi1o,   wsi1of,   64,  4,  2, inv_c);
  repack_b_kernel<<<2, 128, 0, stream>>>(Wpost1e, wpost1ef, 64,  4,  2, inv_c);

  const int ntiles = (N + 15) / 16;
  pre_kernel<<<(ntiles + 3) / 4, 128, 0, stream>>>(node, wp0f, wp1f, h0, h1, N);
  node_gather_kernel<<<(N + 7) / 8, 256, 0, stream>>>(dist, eemb, eidx, freqs,
                                                      W1, b1, W2, b2, W3, b3,
                                                      h0, h1, ioffs, ielist,
                                                      s0, s1o, s1e, N, E);
  post_kernel<<<(ntiles + 3) / 4, 128, 0, stream>>>(node, s0, s1o, s1e,
                                                    wpost0f, wpost1of, wpost1ef,
                                                    wsi0f, wsi1of, (float*)d_out, N);
}
